// MambaSkipConnection_65360812311157
// MI455X (gfx1250) — compile-verified, matrix operand order checked by reading, not run on hardware
//
#include <hip/hip_runtime.h>
#include <hip/hip_bf16.h>

typedef __attribute__((ext_vector_type(16))) _Float16 v16h;
typedef __attribute__((ext_vector_type(8)))  float    v8f;
typedef __attribute__((ext_vector_type(8)))  _Float16 h8;

#define Bb 4
#define Cc_ 192
#define Hh 64
#define Ww 64
#define Ll 4096            // H*W
#define TOK 16384          // B*L
#define DI 384
#define NST 16
#define DTR 12
#define EPSV 1e-5f

// ---------------------------------------------------------------------------
// WMMA fragment load: 16-bit A/B fragment from a row-major (rows, K) matrix.
// Per CDNA5 ISA 16-bit A layout: lane holds row = (given), element j:
//   K = k0 + 8*(lane>>4) + j          (j = 0..7)
//   K = k0 + 8*(lane>>4) + 16 + (j-8) (j = 8..15)
// i.e. two contiguous 16-byte chunks. B fragment uses identical mapping with
// row = n (weights stored (N, K) row-major).
// ---------------------------------------------------------------------------
__device__ __forceinline__ v16h frag_load(const _Float16* p, int ld, int row,
                                          int k0, int lane) {
  const _Float16* base = p + (size_t)row * ld + k0 + ((lane >> 4) << 3);
  h8 lo = *(const h8*)(base);
  h8 hi = *(const h8*)(base + 16);
  v16h r;
#pragma unroll
  for (int i = 0; i < 8; ++i) { r[i] = lo[i]; r[i + 8] = hi[i]; }
  return r;
}

__device__ __forceinline__ v8f wmma_f16(v16h a, v16h b, v8f c) {
  return __builtin_amdgcn_wmma_f32_16x16x32_f16(false, a, false, b,
                                                (short)0, c, false, false);
}

__device__ __forceinline__ float siluf(float x) {
  return x / (1.f + __expf(-x));
}
__device__ __forceinline__ float softplusf(float x) {
  return fmaxf(x, 0.f) + log1pf(__expf(-fabsf(x)));
}

// ---------------------------------------------------------------------------
// Weight prep kernels (fp32 -> fp16, padding, transpose)
// ---------------------------------------------------------------------------
__global__ void k_cvt_f16(const float* __restrict__ s, _Float16* __restrict__ d, int n) {
  int g = blockIdx.x * blockDim.x + threadIdx.x;
  if (g < n) d[g] = (_Float16)s[g];
}

__global__ void k_pad_xproj(const float* __restrict__ s /*44x384*/,
                            _Float16* __restrict__ d /*48x384*/) {
  int g = blockIdx.x * blockDim.x + threadIdx.x;
  if (g >= 48 * DI) return;
  int row = g / DI;
  d[g] = (row < (DTR + 2 * NST)) ? (_Float16)s[g] : (_Float16)0.f;
}

__global__ void k_transpose_rf(const float* __restrict__ rf /*(192,192,3,3)*/,
                               _Float16* __restrict__ wt /*(9,192,192)*/) {
  int g = blockIdx.x * blockDim.x + threadIdx.x;
  if (g >= Cc_ * Cc_ * 9) return;
  int co = g / (Cc_ * 9);
  int r  = g - co * (Cc_ * 9);
  int ci = r / 9;
  int tap = r - ci * 9;
  wt[((size_t)tap * Cc_ + co) * Cc_ + ci] = (_Float16)rf[g];
}

// ---------------------------------------------------------------------------
// LayerNorm over channels; x (B,C,H,W) f32 -> tn (token, C) f16, token-major.
// One block = one (b, y) row of 64 tokens; LDS-tiled for coalesced loads.
// ---------------------------------------------------------------------------
__global__ void k_layernorm(const float* __restrict__ x,
                            const float* __restrict__ lw,
                            const float* __restrict__ lb,
                            _Float16* __restrict__ tn) {
  __shared__ float tile[Cc_][64];
  __shared__ float psum[4][64];
  __shared__ float psq[4][64];
  __shared__ float smu[64], srs[64];
  int by = blockIdx.x;
  int b = by >> 6, y = by & 63;
  const float* xb = x + (size_t)b * Cc_ * Ll + y * Ww;
  for (int i = threadIdx.x; i < Cc_ * 64; i += 256) {
    int c = i >> 6, xx = i & 63;
    tile[c][xx] = xb[(size_t)c * Ll + xx];
  }
  __syncthreads();
  int tok = threadIdx.x & 63, pg = threadIdx.x >> 6; // 4 partials per token
  float s = 0.f, ss = 0.f;
#pragma unroll 4
  for (int j = 0; j < 48; ++j) {
    float v = tile[pg * 48 + j][tok];
    s += v; ss += v * v;
  }
  psum[pg][tok] = s; psq[pg][tok] = ss;
  __syncthreads();
  if (pg == 0) {
    float su = psum[0][tok] + psum[1][tok] + psum[2][tok] + psum[3][tok];
    float sq = psq[0][tok] + psq[1][tok] + psq[2][tok] + psq[3][tok];
    float mu = su * (1.f / Cc_);
    float var = sq * (1.f / Cc_) - mu * mu;
    smu[tok] = mu;
    srs[tok] = rsqrtf(var + EPSV);
  }
  __syncthreads();
  float mu = smu[tok], rs = srs[tok];
  size_t tbase = ((size_t)(b * Ll + y * Ww + tok)) * Cc_;
  for (int j = 0; j < 48; ++j) {
    int c = pg * 48 + j;
    float v = (tile[c][tok] - mu) * rs * lw[c] + lb[c];
    tn[tbase + c] = (_Float16)v;
  }
}

// ---------------------------------------------------------------------------
// in_proj: xz[token][e] = sum_c tn[token][c] * W[e][c]
// M=16384, N=768, K=192; block 256 = 8 waves, one 16x16 tile/wave.
// ---------------------------------------------------------------------------
__global__ void k_gemm_inproj(const _Float16* __restrict__ tn,
                              const _Float16* __restrict__ W,
                              float* __restrict__ xz) {
  int lane = threadIdx.x & 31, wave = threadIdx.x >> 5;
  int mtile = blockIdx.x;
  int ntile = blockIdx.y * 8 + wave;
  int half = lane >> 4;
  int mrow = mtile * 16 + (lane & 15);
  int nrow = ntile * 16 + (lane & 15);
  v8f acc = {};
#pragma unroll
  for (int ks = 0; ks < 6; ++ks) {
    v16h a = frag_load(tn, Cc_, mrow, ks * 32, lane);
    v16h bfr = frag_load(W, Cc_, nrow, ks * 32, lane);
    acc = wmma_f16(a, bfr, acc);
  }
  int n = ntile * 16 + (lane & 15);
#pragma unroll
  for (int r = 0; r < 8; ++r) {
    int m = mtile * 16 + r + 8 * half;
    xz[(size_t)m * (2 * DI) + n] = acc[r];
  }
}

// ---------------------------------------------------------------------------
// Causal depthwise conv1d (k=4) + bias + SiLU on xi (first DI cols of xz).
// ---------------------------------------------------------------------------
__global__ void k_conv1d(const float* __restrict__ xz,
                         const float* __restrict__ cw,
                         const float* __restrict__ cb,
                         _Float16* __restrict__ xt) {
  int g = blockIdx.x * blockDim.x + threadIdx.x;
  if (g >= TOK * DI) return;
  int token = g / DI;
  int d = g - token * DI;
  int b = token >> 12, l = token & (Ll - 1);
  float acc = cb[d];
#pragma unroll
  for (int j = 0; j < 4; ++j) {
    int ls = l - 3 + j;
    if (ls >= 0) acc += cw[d * 4 + j] * xz[((size_t)(b * Ll + ls)) * (2 * DI) + d];
  }
  xt[g] = (_Float16)siluf(acc);
}

// ---------------------------------------------------------------------------
// x_proj: dbl[token][e] = sum_d xt[token][d] * Wx[e][d]   (N padded 44->48)
// M=16384, N=48, K=384; block 96 = 3 waves.
// ---------------------------------------------------------------------------
__global__ void k_gemm_xproj(const _Float16* __restrict__ xt,
                             const _Float16* __restrict__ Wx,
                             float* __restrict__ dbl) {
  int lane = threadIdx.x & 31, wave = threadIdx.x >> 5;
  int mtile = blockIdx.x;
  int ntile = wave;
  int half = lane >> 4;
  int mrow = mtile * 16 + (lane & 15);
  int nrow = ntile * 16 + (lane & 15);
  v8f acc = {};
#pragma unroll
  for (int ks = 0; ks < 12; ++ks) {
    v16h a = frag_load(xt, DI, mrow, ks * 32, lane);
    v16h bfr = frag_load(Wx, DI, nrow, ks * 32, lane);
    acc = wmma_f16(a, bfr, acc);
  }
  int n = ntile * 16 + (lane & 15);
#pragma unroll
  for (int r = 0; r < 8; ++r) {
    int m = mtile * 16 + r + 8 * half;
    dbl[(size_t)m * 48 + n] = acc[r];
  }
}

// ---------------------------------------------------------------------------
// Selective scan, fused: dt_proj + softplus, h-state recurrence (16 states in
// regs), + u*D, * silu(z). One thread per (b, d); 12 blocks x 128 threads.
// ---------------------------------------------------------------------------
__global__ void k_scan(const float* __restrict__ dbl,
                       const _Float16* __restrict__ xt,
                       const float* __restrict__ xz,
                       const float* __restrict__ dtw,
                       const float* __restrict__ dtb,
                       const float* __restrict__ A_log,
                       const float* __restrict__ Dp,
                       _Float16* __restrict__ yact) {
  __shared__ float sh[48];
  int b = blockIdx.x / 3;
  int db = blockIdx.x - b * 3;
  int d = db * 128 + threadIdx.x;

  float w[DTR];
#pragma unroll
  for (int r = 0; r < DTR; ++r) w[r] = dtw[d * DTR + r];
  float bias = dtb[d];
  float Av[NST];
#pragma unroll
  for (int n = 0; n < NST; ++n) Av[n] = -__expf(A_log[d * NST + n]);
  float Dd = Dp[d];
  float h[NST];
#pragma unroll
  for (int n = 0; n < NST; ++n) h[n] = 0.f;

  for (int t = 0; t < Ll; ++t) {
    size_t token = (size_t)b * Ll + t;
    __syncthreads();
    if (threadIdx.x < 48) sh[threadIdx.x] = dbl[token * 48 + threadIdx.x];
    __syncthreads();
    float acc = bias;
#pragma unroll
    for (int r = 0; r < DTR; ++r) acc += sh[r] * w[r];
    float delta = softplusf(acc);
    float u = (float)xt[token * DI + d];
    float du = delta * u;
    float y = 0.f;
#pragma unroll
    for (int n = 0; n < NST; ++n) {
      float dA = __expf(delta * Av[n]);
      h[n] = dA * h[n] + du * sh[DTR + n];
      y += h[n] * sh[DTR + NST + n];
    }
    float z = xz[token * (2 * DI) + DI + d];
    float out = (y + u * Dd) * siluf(z);
    yact[token * DI + d] = (_Float16)out;
  }
}

// ---------------------------------------------------------------------------
// out_proj + residual: t2[token][c] = x[b,c,pos] + sum_d yact[token][d]*Wo[c][d]
// M=16384, N=192, K=384; block 128 = 4 waves, grid.y = 3.
// ---------------------------------------------------------------------------
__global__ void k_gemm_outproj(const _Float16* __restrict__ yact,
                               const _Float16* __restrict__ Wo,
                               const float* __restrict__ x,
                               _Float16* __restrict__ t2) {
  int lane = threadIdx.x & 31, wave = threadIdx.x >> 5;
  int mtile = blockIdx.x;
  int ntile = blockIdx.y * 4 + wave;
  int half = lane >> 4;
  int mrow = mtile * 16 + (lane & 15);
  int nrow = ntile * 16 + (lane & 15);
  v8f acc = {};
#pragma unroll
  for (int ks = 0; ks < 12; ++ks) {
    v16h a = frag_load(yact, DI, mrow, ks * 32, lane);
    v16h bfr = frag_load(Wo, DI, nrow, ks * 32, lane);
    acc = wmma_f16(a, bfr, acc);
  }
  int c = ntile * 16 + (lane & 15);
#pragma unroll
  for (int r = 0; r < 8; ++r) {
    int m = mtile * 16 + r + 8 * half;
    int b = m >> 12, pos = m & (Ll - 1);
    float v = acc[r] + x[((size_t)(b * Cc_ + c)) * Ll + pos];
    t2[(size_t)m * Cc_ + c] = (_Float16)v;
  }
}

// ---------------------------------------------------------------------------
// conv3x3 SAME as implicit WMMA GEMM, A = transposed weights (M = cout),
// B = NHWC pixels (N = 16 consecutive x). Fused bias + BN(eval) + ReLU.
// 54 WMMAs per tile (9 taps x K=192). Border lanes use zero B-fragments.
// ---------------------------------------------------------------------------
__global__ void k_conv3x3(const _Float16* __restrict__ t2,
                          const _Float16* __restrict__ Wt,   // (9,192,192)
                          const float* __restrict__ rfb,
                          const float* __restrict__ bnw,
                          const float* __restrict__ bnb,
                          const float* __restrict__ bnm,
                          const float* __restrict__ bnv,
                          float* __restrict__ out) {
  int lane = threadIdx.x & 31, wave = threadIdx.x >> 5;
  int half = lane >> 4;
  int ptile = blockIdx.x;                 // 1024 tiles of 16 pixels
  int cotile = blockIdx.y * 4 + wave;     // 12 cout tiles
  int b  = ptile >> 8;
  int y  = (ptile >> 2) & 63;
  int x0 = (ptile & 3) << 4;
  int px = x0 + (lane & 15);              // this lane's pixel column (N)
  int co_row = cotile * 16 + (lane & 15); // A-fragment row (M)
  v8f acc = {};
#pragma unroll
  for (int tap = 0; tap < 9; ++tap) {
    int ky = tap / 3, kx = tap - ky * 3;
    int yy = y + ky - 1;
    int xx = px + kx - 1;
    bool ok = (yy >= 0) && (yy < Hh) && (xx >= 0) && (xx < Ww);
    const _Float16* src = t2 + ((size_t)(b * Ll + yy * Ww + xx)) * Cc_;
    const _Float16* wt = Wt + (size_t)tap * Cc_ * Cc_;
#pragma unroll
    for (int ks = 0; ks < 6; ++ks) {
      v16h a = frag_load(wt, Cc_, co_row, ks * 32, lane);
      v16h bfr = {};
      if (ok) {
        const _Float16* p = src + ks * 32 + (half << 3);
        h8 lo = *(const h8*)p;
        h8 hi = *(const h8*)(p + 16);
#pragma unroll
        for (int i = 0; i < 8; ++i) { bfr[i] = lo[i]; bfr[i + 8] = hi[i]; }
      }
      acc = wmma_f16(a, bfr, acc);
    }
  }
#pragma unroll
  for (int r = 0; r < 8; ++r) {
    int co = cotile * 16 + r + 8 * half;
    float v = acc[r] + rfb[co];
    v = (v - bnm[co]) * rsqrtf(bnv[co] + EPSV) * bnw[co] + bnb[co];
    out[((size_t)(b * Cc_ + co)) * Ll + y * Ww + px] = fmaxf(v, 0.f);
  }
}

// ---------------------------------------------------------------------------
// Host orchestration
// ---------------------------------------------------------------------------
extern "C" void kernel_launch(void* const* d_in, const int* in_sizes, int n_in,
                              void* d_out, int out_size, void* d_ws, size_t ws_size,
                              hipStream_t stream) {
  const float* x        = (const float*)d_in[0];
  const float* ln_w     = (const float*)d_in[1];
  const float* ln_b     = (const float*)d_in[2];
  const float* in_w     = (const float*)d_in[3];
  const float* conv_w   = (const float*)d_in[4];
  const float* conv_b   = (const float*)d_in[5];
  const float* xproj_w  = (const float*)d_in[6];
  const float* dtp_w    = (const float*)d_in[7];
  const float* dtp_b    = (const float*)d_in[8];
  const float* A_log    = (const float*)d_in[9];
  const float* Dvec     = (const float*)d_in[10];
  const float* out_w    = (const float*)d_in[11];
  const float* rf_w     = (const float*)d_in[12];
  const float* rf_b     = (const float*)d_in[13];
  const float* bn_w     = (const float*)d_in[14];
  const float* bn_b     = (const float*)d_in[15];
  const float* bn_mean  = (const float*)d_in[16];
  const float* bn_var   = (const float*)d_in[17];
  float* out = (float*)d_out;

  char* ws = (char*)d_ws;
  _Float16* tn   = (_Float16*)(ws + 0);                  //  6,291,456
  float*    xz   = (float*)   (ws + 6291456);            // 50,331,648
  _Float16* xt   = (_Float16*)(ws + 56623104);           // 12,582,912
  float*    dbl  = (float*)   (ws + 69206016);           //  3,145,728
  _Float16* yact = (_Float16*)(ws + 72351744);           // 12,582,912
  _Float16* t2   = (_Float16*)(ws + 84934656);           //  6,291,456
  _Float16* wip  = (_Float16*)(ws + 91226112);           //    294,912
  _Float16* wxp  = (_Float16*)(ws + 91521024);           //     36,864
  _Float16* wop  = (_Float16*)(ws + 91557888);           //    147,456
  _Float16* wrf  = (_Float16*)(ws + 91705344);           //    663,552

  // weight prep
  k_cvt_f16<<<(2 * DI * Cc_ + 255) / 256, 256, 0, stream>>>(in_w, wip, 2 * DI * Cc_);
  k_cvt_f16<<<(Cc_ * DI + 255) / 256, 256, 0, stream>>>(out_w, wop, Cc_ * DI);
  k_pad_xproj<<<(48 * DI + 255) / 256, 256, 0, stream>>>(xproj_w, wxp);
  k_transpose_rf<<<(Cc_ * Cc_ * 9 + 255) / 256, 256, 0, stream>>>(rf_w, wrf);

  // pipeline
  k_layernorm<<<Bb * Hh, 256, 0, stream>>>(x, ln_w, ln_b, tn);
  k_gemm_inproj<<<dim3(TOK / 16, 6), 256, 0, stream>>>(tn, wip, xz);
  k_conv1d<<<(TOK * DI + 255) / 256, 256, 0, stream>>>(xz, conv_w, conv_b, xt);
  k_gemm_xproj<<<dim3(TOK / 16), 96, 0, stream>>>(xt, wxp, dbl);
  k_scan<<<12, 128, 0, stream>>>(dbl, xt, xz, dtp_w, dtp_b, A_log, Dvec, yact);
  k_gemm_outproj<<<dim3(TOK / 16, 3), 128, 0, stream>>>(yact, wop, x, t2);
  k_conv3x3<<<dim3(TOK / 16, 3), 128, 0, stream>>>(t2, wrf, rf_b, bn_w, bn_b,
                                                   bn_mean, bn_var, out);
}
